// encoder_68384469287284
// MI455X (gfx1250) — matrix operand order repaired and checked by reading, not compiled, not run on hardware
//
#include <hip/hip_runtime.h>

typedef __bf16 bf16;
typedef __attribute__((ext_vector_type(16))) __bf16    v16bf;
typedef __attribute__((ext_vector_type(8)))  float     v8f;
typedef __attribute__((ext_vector_type(4)))  unsigned int u32x4;

#define BATCH  64
#define SEQLEN 512
#define EMBED  300
#define EPAD   320          // EMBED padded to multiple of 32 (WMMA K)
#define HID    1024
#define GATES  4096         // 4*HID
#define VOCAB  32000

#define KT_X   (EPAD / 32)          // 10 k-tiles from x@W
#define KT_H   (HID / 32)           // 32 k-tiles from h@R
#define KT_ALL (KT_X + KT_H)        // 42 total
#define KTOT   (EPAD + HID)         // 1344 total K per output column
#define KPAD   1352                 // K padded: col start banks i*36 mod 64, all distinct

#define NBLK   (HID / 16)           // 64 persistent blocks, one H-column tile each
#define SMEM_BYTES (64 * KPAD * 2)  // 4 gates x 16 cols x KPAD bf16 = 173056 B (<320KB WGP LDS)

static __device__ __forceinline__ bf16 f2bf(float f) {
  unsigned int u = __builtin_bit_cast(unsigned int, f);
  u += 0x7FFFu + ((u >> 16) & 1u);              // round-to-nearest-even
  unsigned short hs = (unsigned short)(u >> 16);
  return __builtin_bit_cast(bf16, hs);
}

union AFrag { v16bf v; unsigned int u[8]; };
union BFrag { v16bf v; u32x4 q[2]; };

static __device__ __forceinline__ float sigmoidf_(float x) {
  return 1.0f / (1.0f + __expf(-x));
}

// CDNA5 async copy: per-lane global -> LDS, tracked with ASYNCcnt.
static __device__ __forceinline__ void async_copy16(unsigned lds_byte_off,
                                                    const bf16* src) {
  unsigned long long ga = (unsigned long long)src;
  asm volatile("global_load_async_to_lds_b128 %0, %1, off"
               :: "v"(lds_byte_off), "v"(ga)
               : "memory");
}

static __device__ __forceinline__ void wait_async0() {
#if defined(__has_builtin) && __has_builtin(__builtin_amdgcn_s_wait_asynccnt)
  __builtin_amdgcn_s_wait_asynccnt(0);
#else
  asm volatile("s_wait_asynccnt 0x0" ::: "memory");
#endif
}

// ---------------------------------------------------------------------------
// Embedding gather + cast to bf16, padded E: xbf[b][t][EPAD]
// ---------------------------------------------------------------------------
__global__ void gather_cast_kernel(const float* __restrict__ table,
                                   const int* __restrict__ seq,
                                   bf16* __restrict__ xbf) {
  size_t i = (size_t)blockIdx.x * blockDim.x + threadIdx.x;
  if (i >= (size_t)BATCH * SEQLEN * EPAD) return;
  int e = (int)(i % EPAD);
  size_t bt = i / EPAD;                       // b*SEQLEN + t
  float v = 0.0f;
  if (e < EMBED) {
    int tok = seq[bt];
    v = table[(size_t)tok * EMBED + e];
  }
  xbf[i] = f2bf(v);
}

// Transpose+cast input kernel W[E][G] -> Wt[G][EPAD] (K-contiguous, zero pad)
__global__ void wcast_kernel(const float* __restrict__ W, bf16* __restrict__ Wt) {
  size_t i = (size_t)blockIdx.x * blockDim.x + threadIdx.x;
  if (i >= (size_t)GATES * EPAD) return;
  int e = (int)(i % EPAD);
  size_t g = i / EPAD;
  float v = (e < EMBED) ? W[(size_t)e * GATES + g] : 0.0f;
  Wt[i] = f2bf(v);
}

// Transpose+cast recurrent kernel R[H][G] -> Rt[G][H] (K-contiguous)
__global__ void rcast_kernel(const float* __restrict__ R, bf16* __restrict__ Rt) {
  size_t i = (size_t)blockIdx.x * blockDim.x + threadIdx.x;
  if (i >= (size_t)GATES * HID) return;
  int k = (int)(i % HID);
  size_t g = i / HID;
  Rt[i] = f2bf(R[(size_t)k * GATES + g]);
}

__global__ void init_state_kernel(bf16* __restrict__ h0,
                                  float* __restrict__ hf,
                                  float* __restrict__ cf,
                                  unsigned* __restrict__ bar) {
  int i = blockIdx.x * blockDim.x + threadIdx.x;
  if (i < BATCH * HID) { h0[i] = f2bf(0.0f); hf[i] = 0.0f; cf[i] = 0.0f; }
  if (i == 0) *bar = 0u;
}

__global__ void copy_out_kernel(const float* __restrict__ hf, float* __restrict__ out) {
  int i = blockIdx.x * blockDim.x + threadIdx.x;
  if (i < BATCH * HID) out[i] = hf[i];
}

// ---------------------------------------------------------------------------
// Persistent LSTM kernel: one launch runs all 512 timesteps.
//   grid  = 64 blocks (one 16-wide H-column tile each, covering all 4 gates)
//   block = 128 threads = 4 waves, wave w owns batch rows [16w, 16w+16)
// The block's full weight panel (4 gates x 16 cols x 1344 K bf16 = 168 KB)
// is async-prefilled into LDS ONCE (CDNA5 320KB WGP LDS, ASYNCcnt) and stays
// weight-stationary across all steps. Steps are separated by a device-scope
// atomic grid barrier; h ping-pongs between two global bf16 buffers.
// ---------------------------------------------------------------------------
__global__ __launch_bounds__(128) void lstm_persist_kernel(
    const bf16* __restrict__ xbf,   // [B][T][EPAD]
    const bf16* __restrict__ Wt,    // [G][EPAD]  (K-contiguous per output col)
    const bf16* __restrict__ Rt,    // [G][H]     (K-contiguous per output col)
    const float* __restrict__ bias, // [G]
    const int* __restrict__ seq,    // [B][T]
    bf16* __restrict__ hb0,         // [B][H] bf16 ping
    bf16* __restrict__ hb1,         // [B][H] bf16 pong
    float* __restrict__ hf,         // [B][H] f32 state
    float* __restrict__ cf,         // [B][H] f32 state
    unsigned* __restrict__ bar) {   // grid-barrier arrival counter
  extern __shared__ bf16 smem[];            // [64 cols][KPAD]

  const int tid   = threadIdx.x;
  const int lane  = tid & 31;
  const int wave  = tid >> 5;
  const int m0    = wave << 4;              // batch-row tile base
  const int n0    = blockIdx.x << 4;        // H-column tile base
  const int lrow  = lane & 15;
  const int khalf = lane >> 4;              // A frag: which K-half this lane holds
  const int ksub  = khalf << 3;             // B frag: K element offset 0 or 8

  // ---- one-time weight panel prefill: 10752 16B chunks, 84 per thread ----
  {
    const int CHK_PER_COL = KTOT / 8;       // 168
    for (int c = tid; c < 64 * CHK_PER_COL; c += 128) {
      int colIdx = c / CHK_PER_COL;         // 0..63 = gi*16 + col
      int kc     = c % CHK_PER_COL;
      int gi     = colIdx >> 4;
      int col    = colIdx & 15;
      int kk     = kc * 8;
      const bf16* src = (kk < EPAD)
          ? Wt + (size_t)(gi * HID + n0 + col) * EPAD + kk
          : Rt + (size_t)(gi * HID + n0 + col) * HID + (kk - EPAD);
      unsigned lo = (unsigned)(size_t)&smem[(size_t)colIdx * KPAD + kk];
      async_copy16(lo, src);
    }
    wait_async0();
    __syncthreads();
  }

  // A-fragment per-lane K offsets within a 32-wide K tile (ISA 16-bit A 16x32)
  int kofs[8];
#pragma unroll
  for (int i = 0; i < 8; ++i)
    kofs[i] = ((i & 3) << 1) + ((i >> 2) << 4) + (khalf << 3);

  // hoisted epilogue constants
  const int ncol = n0 + lrow;
  const float bi  = bias[ncol];
  const float bff = bias[HID + ncol];
  const float bg  = bias[2 * HID + ncol];
  const float bo  = bias[3 * HID + ncol];
  const int rbase = m0 + (khalf << 3);      // lanes>=16 hold rows M=8..15

  const bf16* sbase = smem + (size_t)lrow * KPAD + ksub;   // per-lane B base

  for (int t = 0; t < SEQLEN; ++t) {
    const bf16* hin  = (t & 1) ? hb1 : hb0;
    bf16*       hout = (t & 1) ? hb0 : hb1;
    const bf16* xrow = xbf + ((size_t)(m0 + lrow) * SEQLEN + t) * EPAD;
    const bf16* hrow = hin + (size_t)(m0 + lrow) * HID;

    v8f acc[4] = {{}, {}, {}, {}};

    auto mma_tile = [&](const bf16* ar, int ktAbs) {
      AFrag a;
#pragma unroll
      for (int i = 0; i < 8; ++i)
        a.u[i] = *(const unsigned int*)(ar + kofs[i]);
      const bf16* sb = sbase + ktAbs * 32;
      BFrag b[4];
#pragma unroll
      for (int gi = 0; gi < 4; ++gi) {
        b[gi].q[0] = *(const u32x4*)(sb + (size_t)gi * 16 * KPAD);
        b[gi].q[1] = *(const u32x4*)(sb + (size_t)gi * 16 * KPAD + 16);
      }
#pragma unroll
      for (int gi = 0; gi < 4; ++gi)
        acc[gi] = __builtin_amdgcn_wmma_f32_16x16x32_bf16(
            false, a.v, false, b[gi].v, (short)0, acc[gi], false, false);
    };

    for (int kt = 0; kt < KT_X; ++kt) mma_tile(xrow + kt * 32, kt);
    for (int kt = 0; kt < KT_H; ++kt) mma_tile(hrow + kt * 32, KT_X + kt);

    // ---------------- gate math + state update ----------------
#pragma unroll
    for (int r = 0; r < 8; ++r) {
      const int row = rbase + r;
      const size_t idx = (size_t)row * HID + ncol;
      float ig = sigmoidf_(acc[0][r] + bi);
      float fg = sigmoidf_(acc[1][r] + bff);
      float gg = tanhf(acc[2][r] + bg);
      float og = sigmoidf_(acc[3][r] + bo);
      float c_old = cf[idx];
      float c_new = fg * c_old + ig * gg;
      float h_new = og * tanhf(c_new);
      bool m = (seq[(size_t)row * SEQLEN + t] != 0);
      float h_o = m ? h_new : hf[idx];
      float c_o = m ? c_new : c_old;
      hf[idx] = h_o;
      cf[idx] = c_o;
      hout[idx] = f2bf(h_o);
    }

    // ---------------- device-scope grid barrier ----------------
    __threadfence();                        // release this block's h stores
    __syncthreads();
    if (tid == 0) {
      __hip_atomic_fetch_add(bar, 1u, __ATOMIC_ACQ_REL, __HIP_MEMORY_SCOPE_AGENT);
      const unsigned target = (unsigned)(t + 1) * (unsigned)NBLK;
      while (__hip_atomic_load(bar, __ATOMIC_ACQUIRE, __HIP_MEMORY_SCOPE_AGENT) < target)
        __builtin_amdgcn_s_sleep(2);
    }
    __syncthreads();
    __threadfence();                        // acquire other blocks' h stores
  }
}

// ---------------------------------------------------------------------------
extern "C" void kernel_launch(void* const* d_in, const int* in_sizes, int n_in,
                              void* d_out, int out_size, void* d_ws, size_t ws_size,
                              hipStream_t stream) {
  const float* table = (const float*)d_in[0];  // [VOCAB][EMBED]
  const float* W     = (const float*)d_in[1];  // [EMBED][GATES]
  const float* R     = (const float*)d_in[2];  // [HID][GATES]
  const float* bias  = (const float*)d_in[3];  // [GATES]
  const int*   seq   = (const int*)d_in[4];    // [BATCH][SEQLEN]

  char* ws = (char*)d_ws;
  size_t off = 0;
  auto carve = [&](size_t bytes) -> void* {
    void* p = ws + off;
    off = (off + bytes + 255) & ~(size_t)255;
    return p;
  };
  bf16*  xbf   = (bf16*)carve((size_t)BATCH * SEQLEN * EPAD * sizeof(bf16));
  bf16*  Wt    = (bf16*)carve((size_t)GATES * EPAD * sizeof(bf16));
  bf16*  Rt    = (bf16*)carve((size_t)GATES * HID * sizeof(bf16));
  bf16*  hbuf0 = (bf16*)carve((size_t)BATCH * HID * sizeof(bf16));
  bf16*  hbuf1 = (bf16*)carve((size_t)BATCH * HID * sizeof(bf16));
  float* hf    = (float*)carve((size_t)BATCH * HID * sizeof(float));
  float* cf    = (float*)carve((size_t)BATCH * HID * sizeof(float));
  unsigned* bar = (unsigned*)carve(256);
  (void)ws_size; (void)in_sizes; (void)n_in; (void)out_size;

  {
    size_t n = (size_t)BATCH * SEQLEN * EPAD;
    gather_cast_kernel<<<(unsigned)((n + 255) / 256), 256, 0, stream>>>(table, seq, xbf);
  }
  {
    size_t n = (size_t)GATES * EPAD;
    wcast_kernel<<<(unsigned)((n + 255) / 256), 256, 0, stream>>>(W, Wt);
  }
  {
    size_t n = (size_t)GATES * HID;
    rcast_kernel<<<(unsigned)((n + 255) / 256), 256, 0, stream>>>(R, Rt);
  }
  init_state_kernel<<<(BATCH * HID + 255) / 256, 256, 0, stream>>>(hbuf0, hf, cf, bar);

  lstm_persist_kernel<<<NBLK, 128, SMEM_BYTES, stream>>>(
      xbf, Wt, Rt, bias, seq, hbuf0, hbuf1, hf, cf, bar);

  copy_out_kernel<<<(BATCH * HID + 255) / 256, 256, 0, stream>>>(hf, (float*)d_out);
}
